// MambaBlock_8796093022910
// MI455X (gfx1250) — compile-verified, matrix operand order checked by reading, not run on hardware
//
#include <hip/hip_runtime.h>
#include <hip/hip_bf16.h>

// ---------------------------------------------------------------------------
// MambaBlock for MI455X (gfx1250): bf16 WMMA GEMMs + fused scan/conv.
// B=8, T=2048, D=512, INNER=1024, K_CONV=4.  M = B*T = 16384 rows.
// Each wave computes a 16(M) x 64(N) output tile: one A fragment feeds 4
// v_wmma_f32_16x16x32_bf16 accumulators (register-level A reuse).
// ---------------------------------------------------------------------------

#define MB_B      8
#define MB_T      2048
#define MB_D      512
#define MB_INNER  1024
#define MB_M      (MB_B * MB_T)        // 16384
#define MB_N_IN   (3 * MB_INNER)       // 3072
#define LN_EPS    1e-5f

typedef __bf16 v16bf __attribute__((ext_vector_type(16)));
typedef __bf16 v8bf  __attribute__((ext_vector_type(8)));
typedef float  v8f   __attribute__((ext_vector_type(8)));

// ---------------------------------------------------------------------------
// fp32 -> bf16 conversion kernel (weights)
// ---------------------------------------------------------------------------
__global__ __launch_bounds__(256) void mb_cvt_bf16(const float* __restrict__ in,
                                                   __bf16* __restrict__ out, int n) {
    int i = blockIdx.x * blockDim.x + threadIdx.x;
    if (i < n) out[i] = (__bf16)in[i];
}

// ---------------------------------------------------------------------------
// LayerNorm: one wave32 per row of 512, writes bf16 xn.
// ---------------------------------------------------------------------------
__global__ __launch_bounds__(256) void mb_ln(const float* __restrict__ x,
                                             const float* __restrict__ w,
                                             const float* __restrict__ b,
                                             __bf16* __restrict__ xn) {
    const int wave = threadIdx.x >> 5;
    const int lane = threadIdx.x & 31;
    const int row  = blockIdx.x * (blockDim.x >> 5) + wave;
    if (row >= MB_M) return;
    const float* xr = x + (size_t)row * MB_D;

    float v[16];
    float s = 0.f;
#pragma unroll
    for (int i = 0; i < 16; ++i) { v[i] = xr[lane + 32 * i]; s += v[i]; }
#pragma unroll
    for (int off = 16; off >= 1; off >>= 1) s += __shfl_xor(s, off, 32);
    const float mu = s * (1.f / MB_D);

    float ss = 0.f;
#pragma unroll
    for (int i = 0; i < 16; ++i) { float d = v[i] - mu; ss += d * d; }
#pragma unroll
    for (int off = 16; off >= 1; off >>= 1) ss += __shfl_xor(ss, off, 32);
    const float inv = rsqrtf(ss * (1.f / MB_D) + LN_EPS);

    __bf16* xo = xn + (size_t)row * MB_D;
#pragma unroll
    for (int i = 0; i < 16; ++i) {
        const int col = lane + 32 * i;
        xo[col] = (__bf16)((v[i] - mu) * inv * w[col] + b[col]);
    }
}

// ---------------------------------------------------------------------------
// WMMA mainloop: one wave computes a 16x64 f32 tile of A[M,K] * W[N,K]^T.
// A fragment (16-bit A layout: lane&15 = M row, lane half h holds K 8h..8h+7, 16+8h..23+8h) is
// loaded once per K-step and reused by 4 WMMAs against 4 B fragments.
// W is [N,K] row-major == B^T, so B fragments are contiguous-K loads too.
// ---------------------------------------------------------------------------
__device__ __forceinline__ void mb_wmma_tile4(const __bf16* __restrict__ A,
                                              const __bf16* __restrict__ W,
                                              int row_base, int col_base, int K,
                                              v8f acc[4]) {
    const int lane = threadIdx.x & 31;
    const int l15  = lane & 15;
    const int ksel = (lane & 16) ? 8 : 0;
    const __bf16* ar = A + (size_t)(row_base + l15) * K + ksel;
    const __bf16* br = W + (size_t)(col_base + l15) * K + ksel;
    const size_t bstride = (size_t)16 * K;   // next 16-col B tile

#pragma unroll 2
    for (int k0 = 0; k0 < K; k0 += 32) {
        __builtin_prefetch(ar + k0 + 128, 0, 3);   // high-locality prefetch
        v8bf alo = *(const v8bf*)(ar + k0);
        v8bf ahi = *(const v8bf*)(ar + k0 + 16);
        v16bf a;
#pragma unroll
        for (int i = 0; i < 8; ++i) { a[i] = alo[i]; a[i + 8] = ahi[i]; }

#pragma unroll
        for (int j = 0; j < 4; ++j) {
            const __bf16* bj = br + (size_t)j * bstride + k0;
            __builtin_prefetch(bj + 128, 0, 3);
            v8bf blo = *(const v8bf*)(bj);
            v8bf bhi = *(const v8bf*)(bj + 16);
            v16bf bb;
#pragma unroll
            for (int i = 0; i < 8; ++i) { bb[i] = blo[i]; bb[i + 8] = bhi[i]; }
            acc[j] = __builtin_amdgcn_wmma_f32_16x16x32_bf16(
                /*neg_a=*/false, a, /*neg_b=*/false, bb,
                /*c_mod=*/(short)0, acc[j], /*reuse_a=*/false, /*reuse_b=*/false);
        }
    }
}

// C/D layout: lane&15 = N col; VGPR v = row (v + (lane&16 ? 8 : 0)).

// ---------------------------------------------------------------------------
// GEMM 1: h = xn @ in_proj_w^T + b, fused split + SiLU / sigmoid / tanh.
// Each 64-wide wave tile lies entirely inside one 1024-wide segment, so the
// proj/gate/drive selection is wave-uniform.
// ---------------------------------------------------------------------------
__global__ __launch_bounds__(256) void mb_gemm_inproj(const __bf16* __restrict__ xn,
                                                      const __bf16* __restrict__ wi,
                                                      const float* __restrict__ bias,
                                                      __bf16* __restrict__ proj,
                                                      __bf16* __restrict__ gate,
                                                      __bf16* __restrict__ drive) {
    const int tiles_n = MB_N_IN / 64;                      // 48
    const int tile = blockIdx.x * 8 + (threadIdx.x >> 5);
    const int row_base = (tile / tiles_n) * 16;
    const int col_base = (tile % tiles_n) * 64;

    v8f acc[4] = {};
    mb_wmma_tile4(xn, wi, row_base, col_base, MB_D, acc);

    const int lane = threadIdx.x & 31;
    const int l15  = lane & 15;
    const int moff = (lane & 16) ? 8 : 0;
    const int seg  = col_base >> 10;                       // 0=proj,1=gate,2=drive
    const int nloc0 = (col_base & (MB_INNER - 1)) + l15;

#pragma unroll
    for (int j = 0; j < 4; ++j) {
        const int n    = col_base + j * 16 + l15;
        const int nloc = nloc0 + j * 16;
        const float bv = bias[n];
#pragma unroll
        for (int v = 0; v < 8; ++v) {
            const int m  = row_base + v + moff;
            const float t = acc[j][v] + bv;
            const size_t idx = (size_t)m * MB_INNER + nloc;
            if (seg == 0) {
                const float s = 1.f / (1.f + __expf(-t));          // SiLU
                proj[idx] = (__bf16)(t * s);
            } else if (seg == 1) {
                gate[idx] = (__bf16)(1.f / (1.f + __expf(-t)));    // sigmoid
            } else {
                drive[idx] = (__bf16)tanhf(t);                     // tanh
            }
        }
    }
}

// ---------------------------------------------------------------------------
// GEMM 2: delta = clamp(sigmoid(xn @ delta_w^T + b)); d = delta * drive.
// ---------------------------------------------------------------------------
__global__ __launch_bounds__(256) void mb_gemm_delta(const __bf16* __restrict__ xn,
                                                     const __bf16* __restrict__ wd,
                                                     const float* __restrict__ bias,
                                                     const __bf16* __restrict__ drive,
                                                     __bf16* __restrict__ dbuf) {
    const int tiles_n = MB_INNER / 64;                     // 16
    const int tile = blockIdx.x * 8 + (threadIdx.x >> 5);
    const int row_base = (tile / tiles_n) * 16;
    const int col_base = (tile % tiles_n) * 64;

    v8f acc[4] = {};
    mb_wmma_tile4(xn, wd, row_base, col_base, MB_D, acc);

    const int lane = threadIdx.x & 31;
    const int l15  = lane & 15;
    const int moff = (lane & 16) ? 8 : 0;
#pragma unroll
    for (int j = 0; j < 4; ++j) {
        const int n    = col_base + j * 16 + l15;
        const float bv = bias[n];
#pragma unroll
        for (int v = 0; v < 8; ++v) {
            const int m = row_base + v + moff;
            const float t  = acc[j][v] + bv;
            const float dl = fmaxf(1.f / (1.f + __expf(-t)), 1e-4f);
            const size_t idx = (size_t)m * MB_INNER + n;
            dbuf[idx] = (__bf16)(dl * (float)drive[idx]);
        }
    }
}

// ---------------------------------------------------------------------------
// Causal depthwise conv (K=4, rolling regs) + f32 recurrence + gate/skip.
// One thread per (batch, channel); channels contiguous per t -> coalesced.
// ---------------------------------------------------------------------------
__global__ __launch_bounds__(256) void mb_conv_scan(const __bf16* __restrict__ proj,
                                                    const __bf16* __restrict__ dbuf,
                                                    const __bf16* __restrict__ gate,
                                                    const float* __restrict__ conv_w,
                                                    const float* __restrict__ conv_b,
                                                    const float* __restrict__ state_mix,
                                                    const float* __restrict__ skip_scale,
                                                    __bf16* __restrict__ yin) {
    const int tid = blockIdx.x * blockDim.x + threadIdx.x;   // 8192 threads
    const int b = tid >> 10;
    const int c = tid & (MB_INNER - 1);

    const float w0 = conv_w[c * 4 + 0], w1 = conv_w[c * 4 + 1];
    const float w2 = conv_w[c * 4 + 2], w3 = conv_w[c * 4 + 3];
    const float cb = conv_b[c];
    const float a  = 1.f / (1.f + __expf(-state_mix[c]));
    const float sk = skip_scale[c];

    const size_t base = (size_t)b * MB_T * MB_INNER + c;
    float p1 = 0.f, p2 = 0.f, p3 = 0.f;   // proj[t-1..t-3]
    float state = 0.f;
    for (int t = 0; t < MB_T; ++t) {
        const size_t idx = base + (size_t)t * MB_INNER;
        const float p0 = (float)proj[idx];
        const float conv = cb + w3 * p0 + w2 * p1 + w1 * p2 + w0 * p3;
        state = a * state + (float)dbuf[idx];
        const float y = (state + sk * conv) * (float)gate[idx];
        yin[idx] = (__bf16)y;
        p3 = p2; p2 = p1; p1 = p0;
    }
}

// ---------------------------------------------------------------------------
// GEMM 3: out = residual + yin @ out_proj_w^T + b.
// ---------------------------------------------------------------------------
__global__ __launch_bounds__(256) void mb_gemm_out(const __bf16* __restrict__ yin,
                                                   const __bf16* __restrict__ wo,
                                                   const float* __restrict__ bias,
                                                   const float* __restrict__ residual,
                                                   float* __restrict__ out) {
    const int tiles_n = MB_D / 64;                         // 8
    const int tile = blockIdx.x * 8 + (threadIdx.x >> 5);
    const int row_base = (tile / tiles_n) * 16;
    const int col_base = (tile % tiles_n) * 64;

    v8f acc[4] = {};
    mb_wmma_tile4(yin, wo, row_base, col_base, MB_INNER, acc);

    const int lane = threadIdx.x & 31;
    const int l15  = lane & 15;
    const int moff = (lane & 16) ? 8 : 0;
#pragma unroll
    for (int j = 0; j < 4; ++j) {
        const int n    = col_base + j * 16 + l15;
        const float bv = bias[n];
#pragma unroll
        for (int v = 0; v < 8; ++v) {
            const int m = row_base + v + moff;
            out[(size_t)m * MB_D + n] = residual[(size_t)m * MB_D + n] + acc[j][v] + bv;
        }
    }
}

// ---------------------------------------------------------------------------
// Host launcher
// ---------------------------------------------------------------------------
extern "C" void kernel_launch(void* const* d_in, const int* in_sizes, int n_in,
                              void* d_out, int out_size, void* d_ws, size_t ws_size,
                              hipStream_t stream) {
    (void)in_sizes; (void)n_in; (void)out_size; (void)ws_size;

    const float* x      = (const float*)d_in[0];
    const float* ln_w   = (const float*)d_in[1];
    const float* ln_b   = (const float*)d_in[2];
    const float* in_w   = (const float*)d_in[3];
    const float* in_b   = (const float*)d_in[4];
    const float* conv_w = (const float*)d_in[5];
    const float* conv_b = (const float*)d_in[6];
    const float* dl_w   = (const float*)d_in[7];
    const float* dl_b   = (const float*)d_in[8];
    const float* smix   = (const float*)d_in[9];
    const float* skip   = (const float*)d_in[10];
    const float* out_w  = (const float*)d_in[11];
    const float* out_b  = (const float*)d_in[12];
    float* out = (float*)d_out;

    // Workspace layout (bf16 buffers, all 16B-aligned sizes)
    char* ws = (char*)d_ws;
    size_t off = 0;
    __bf16* xnB    = (__bf16*)(ws + off); off += (size_t)MB_M * MB_D * 2;        // 16 MB
    __bf16* wiB    = (__bf16*)(ws + off); off += (size_t)MB_N_IN * MB_D * 2;     //  3 MB
    __bf16* wdB    = (__bf16*)(ws + off); off += (size_t)MB_INNER * MB_D * 2;    //  1 MB
    __bf16* woB    = (__bf16*)(ws + off); off += (size_t)MB_D * MB_INNER * 2;    //  1 MB
    __bf16* projB  = (__bf16*)(ws + off); off += (size_t)MB_M * MB_INNER * 2;    // 32 MB
    __bf16* gateB  = (__bf16*)(ws + off); off += (size_t)MB_M * MB_INNER * 2;    // 32 MB
    __bf16* driveB = (__bf16*)(ws + off); off += (size_t)MB_M * MB_INNER * 2;    // 32 MB
    __bf16* dB     = (__bf16*)(ws + off); off += (size_t)MB_M * MB_INNER * 2;    // 32 MB
    __bf16* yinB   = (__bf16*)(ws + off); off += (size_t)MB_M * MB_INNER * 2;    // 32 MB

    // 1) Weight conversion to bf16
    {
        int n1 = MB_N_IN * MB_D, n2 = MB_INNER * MB_D, n3 = MB_D * MB_INNER;
        mb_cvt_bf16<<<(n1 + 255) / 256, 256, 0, stream>>>(in_w, wiB, n1);
        mb_cvt_bf16<<<(n2 + 255) / 256, 256, 0, stream>>>(dl_w, wdB, n2);
        mb_cvt_bf16<<<(n3 + 255) / 256, 256, 0, stream>>>(out_w, woB, n3);
    }

    // 2) LayerNorm (wave per row, 8 waves/block)
    mb_ln<<<MB_M / 8, 256, 0, stream>>>(x, ln_w, ln_b, xnB);

    // 3) in_proj GEMM + activations  (1024 x 48 wave tiles, 8/block)
    mb_gemm_inproj<<<(MB_M / 16) * (MB_N_IN / 64) / 8, 256, 0, stream>>>(
        xnB, wiB, in_b, projB, gateB, driveB);

    // 4) delta GEMM + d = delta * drive  (1024 x 16 wave tiles)
    mb_gemm_delta<<<(MB_M / 16) * (MB_INNER / 64) / 8, 256, 0, stream>>>(
        xnB, wdB, dl_b, driveB, dB);

    // 5) depthwise causal conv + recurrence + gate/skip
    mb_conv_scan<<<(MB_B * MB_INNER) / 256, 256, 0, stream>>>(
        projB, dB, gateB, conv_w, conv_b, smix, skip, yinB);

    // 6) out_proj GEMM + residual  (1024 x 8 wave tiles)
    mb_gemm_out<<<(MB_M / 16) * (MB_D / 64) / 8, 256, 0, stream>>>(
        yinB, woB, out_b, x, out);
}
